// TorchAOInt8Linear_91147795955913
// MI455X (gfx1250) — compile-verified, matrix operand order checked by reading, not run on hardware
//
#include <hip/hip_runtime.h>

// ---------------------------------------------------------------------------
// int8 weight-only quantized linear:  y[M,N] = x[M,K] @ (w_q[N,K]*scale[N])^T + bias[N]
// M = B*S = 4096, K = 4096, N = 11008  (fixed by reference)
// f16 WMMA (V_WMMA_F32_16X16X32_F16) with fp32 accumulate.
//   - w_q is EXACT in f16 (|v| <= 127, 10 mantissa bits); scale folded into epilogue
//   - x converted fp32->f16 via v_cvt_pk_rtz_f16_f32 (2 elems / instr)
//   - int8->f16 via perm + packed-f16 magic-number (5 VALU / 4 elems)
// Tile: block 128(M) x 256(N), 8 waves as 2x4, each wave 64x64 (4x4 WMMA tiles)
//   -> 1 ds_read_b128 per WMMA (was 1.5), LDS-read traffic cut by 1.5x.
// ---------------------------------------------------------------------------

typedef __attribute__((ext_vector_type(2)))  _Float16 h2;
typedef __attribute__((ext_vector_type(16))) _Float16 v16h;
typedef __attribute__((ext_vector_type(8)))  float    v8f;

constexpr int Mdim = 4096;
constexpr int Kdim = 4096;
constexpr int Ndim = 11008;

constexpr int BM = 128;
constexpr int BN = 256;
constexpr int BK = 64;
constexpr int LDA = BK + 8;   // padded f16 stride: 72 elems = 144 B -> bank-conflict-free b128 reads
constexpr int LDB = BK + 8;

union FragH {
  v16h  v;
  uint4 q[2];
};

__global__ __launch_bounds__(256)
void int8wo_linear_f16_wmma(const float* __restrict__ x,
                            const signed char* __restrict__ wq,
                            const float* __restrict__ scale,
                            const float* __restrict__ bias,
                            float* __restrict__ out) {
  __shared__ unsigned short As[BM * LDA];   // x tile, f16   (18.4 KB)
  __shared__ unsigned short Ws[BN * LDB];   // w  tile, f16  (36.9 KB)

  const int tid   = threadIdx.x;
  const int lane  = tid & 31;
  const int wave  = tid >> 5;
  const int waveM = wave & 1;    // 2 waves down M  -> 64 rows each
  const int waveN = wave >> 1;   // 4 waves across N -> 64 cols each

  const int tileN = blockIdx.x * BN;
  const int tileM = blockIdx.y * BM;

  // ---- global-load register staging (one BK=64 stage) ----------------------
  // A: 128x64 fp32  = 8 passes of (256 thr * float4)
  // W: 256x64 int8  = 4 passes of (256 thr * 16 bytes)
  const int ar = tid >> 4;          // A: row within 16-row pass
  const int ac = (tid & 15) * 4;    // A: fp32 column
  const int wr = tid >> 2;          // W: row within 64-row pass
  const int wc = (tid & 3) * 16;    // W: int8 column

  float4 aReg[8];
  uint4  wReg[4];

  auto load_stage = [&](int kk) {
#pragma unroll
    for (int p = 0; p < 8; ++p)
      aReg[p] = *(const float4*)(x + (size_t)(tileM + p * 16 + ar) * Kdim + kk + ac);
#pragma unroll
    for (int p = 0; p < 4; ++p)
      wReg[p] = *(const uint4*)(wq + (size_t)(tileN + p * 64 + wr) * Kdim + kk + wc);
  };

  const h2 c1152 = {(_Float16)1152.0f, (_Float16)1152.0f};  // 0x6400 bias + 128

  auto stage_to_lds = [&]() {
    // x tile: packed RTZ f32->f16, 2 elems per instruction
#pragma unroll
    for (int p = 0; p < 8; ++p) {
      uint2 s;
      s.x = __builtin_bit_cast(unsigned, __builtin_amdgcn_cvt_pkrtz(aReg[p].x, aReg[p].y));
      s.y = __builtin_bit_cast(unsigned, __builtin_amdgcn_cvt_pkrtz(aReg[p].z, aReg[p].w));
      *(uint2*)(As + (p * 16 + ar) * LDA + ac) = s;
    }
    // w tile: int8 -> f16 exact, via 0x6400-magic:
    //   b' = b ^ 0x80 in [0,255];  f16(0x6400 | b') = 1024 + b';  minus 1152 -> b
#pragma unroll
    for (int p = 0; p < 4; ++p) {
      const unsigned* src = (const unsigned*)&wReg[p];
      unsigned d[8];
#pragma unroll
      for (int j = 0; j < 4; ++j) {
        const unsigned t  = src[j] ^ 0x80808080u;
        const unsigned lo = __builtin_amdgcn_perm(0x64646464u, t, 0x04010400u); // {64,b1',64,b0'}
        const unsigned hi = __builtin_amdgcn_perm(0x64646464u, t, 0x04030402u); // {64,b3',64,b2'}
        d[2 * j]     = __builtin_bit_cast(unsigned, __builtin_bit_cast(h2, lo) - c1152);
        d[2 * j + 1] = __builtin_bit_cast(unsigned, __builtin_bit_cast(h2, hi) - c1152);
      }
      unsigned short* dst = Ws + (p * 64 + wr) * LDB + wc;
      ((uint4*)dst)[0] = make_uint4(d[0], d[1], d[2], d[3]);
      ((uint4*)dst)[1] = make_uint4(d[4], d[5], d[6], d[7]);
    }
  };

  // ---- accumulators: 4 (M) x 4 (N) 16x16 tiles per wave ---------------------
  v8f acc[4][4] = {};

  const int l15 = lane & 15;
  const int kh  = (lane >> 4) << 3;   // ISA 16-bit A/B layout: hi lanes hold K+8

  auto compute = [&]() {
#pragma unroll
    for (int ks = 0; ks < BK; ks += 32) {
      FragH a[4];
#pragma unroll
      for (int mt = 0; mt < 4; ++mt) {
        const unsigned short* p = As + (waveM * 64 + mt * 16 + l15) * LDA + ks + kh;
        a[mt].q[0] = *(const uint4*)p;         // K  0.. 7 (lo lanes) /  8..15 (hi)
        a[mt].q[1] = *(const uint4*)(p + 16);  // K 16..23 (lo lanes) / 24..31 (hi)
      }
#pragma unroll
      for (int nt = 0; nt < 4; ++nt) {
        FragH b;   // just-in-time B fragment keeps peak VGPR pressure < 256
        const unsigned short* p = Ws + (waveN * 64 + nt * 16 + l15) * LDB + ks + kh;
        b.q[0] = *(const uint4*)p;
        b.q[1] = *(const uint4*)(p + 16);
#pragma unroll
        for (int mt = 0; mt < 4; ++mt)
          acc[mt][nt] = __builtin_amdgcn_wmma_f32_16x16x32_f16(
              false, a[mt].v, false, b.v, (short)0, acc[mt][nt], false, false);
      }
    }
  };

  // ---- main K loop: software-pipelined global->regs->LDS->WMMA --------------
  load_stage(0);
  for (int kk = 0; kk < Kdim; kk += BK) {
    stage_to_lds();
    __syncthreads();
    if (kk + BK < Kdim) {
      load_stage(kk + BK);                      // overlaps with WMMAs below
      if (kk + 2 * BK < Kdim) {                 // warm GL2 two stages ahead
        __builtin_prefetch(x  + (size_t)(tileM + ar) * Kdim + kk + 2 * BK + ac, 0, 0);
        __builtin_prefetch(wq + (size_t)(tileN + wr) * Kdim + kk + 2 * BK + wc, 0, 0);
      }
    }
    compute();
    __syncthreads();
  }

  // ---- epilogue: acc * scale[col] + bias[col] -------------------------------
  // C/D layout: VGPR v -> M = v (lanes 0-15) / v+8 (lanes 16-31); N = lane&15
  const int rsel = (lane >> 4) * 8;
#pragma unroll
  for (int nt = 0; nt < 4; ++nt) {
    const int col = tileN + waveN * 64 + nt * 16 + l15;
    const float sc = scale[col];
    const float bs = bias[col];
#pragma unroll
    for (int mt = 0; mt < 4; ++mt) {
      const int r0 = tileM + waveM * 64 + mt * 16 + rsel;
#pragma unroll
      for (int v = 0; v < 8; ++v)
        out[(size_t)(r0 + v) * Ndim + col] = acc[mt][nt][v] * sc + bs;
    }
  }
}

extern "C" void kernel_launch(void* const* d_in, const int* in_sizes, int n_in,
                              void* d_out, int out_size, void* d_ws, size_t ws_size,
                              hipStream_t stream) {
  (void)in_sizes; (void)n_in; (void)out_size; (void)d_ws; (void)ws_size;
  const float*       x     = (const float*)d_in[0];
  const signed char* wq    = (const signed char*)d_in[1];
  const float*       scale = (const float*)d_in[2];
  const float*       bias  = (const float*)d_in[3];
  float*             out   = (float*)d_out;

  dim3 grid(Ndim / BN, Mdim / BM);   // 43 x 32 = 1376 workgroups
  dim3 block(256);                   // 8 wave32 waves
  int8wo_linear_f16_wmma<<<grid, block, 0, stream>>>(x, wq, scale, bias, out);
}
